// PRSlot_Moule_34144990003588
// MI455X (gfx1250) — matrix operand order repaired and checked by reading, not compiled, not run on hardware
//
#include <hip/hip_runtime.h>
#include <hip/hip_bf16.h>
#include <math.h>

// ---------------------------------------------------------------------------
// PRSlot module for MI455X (gfx1250, wave32, WMMA, LDS-staged GEMMs).
// Shapes: N=8, Cin=512, T=192, hid=256, key=128, ADJ=(1,3,5), GROUPS=32.
// d_out = [ xs@A : 8*256*192 f32 | A : 8*192*192 f32 ]
// ---------------------------------------------------------------------------

typedef __attribute__((ext_vector_type(16))) _Float16 v16h;
typedef __attribute__((ext_vector_type(8)))  _Float16 h8;
typedef __attribute__((ext_vector_type(8)))  float    v8f;

#define TT 192
#define NB 8

__device__ inline v16h frag16(h8 lo, h8 hi) {
  v16h r;
#pragma unroll
  for (int e = 0; e < 8; ++e) { r[e] = lo[e]; r[8 + e] = hi[e]; }
  return r;
}

__device__ inline h8 cvt8(float4 f0, float4 f1) {
  h8 h;
  h[0] = (_Float16)f0.x; h[1] = (_Float16)f0.y;
  h[2] = (_Float16)f0.z; h[3] = (_Float16)f0.w;
  h[4] = (_Float16)f1.x; h[5] = (_Float16)f1.y;
  h[6] = (_Float16)f1.z; h[7] = (_Float16)f1.w;
  return h;
}

// ---------------------------------------------------------------------------
// LDS-staged GEMM: C[n] = A[n] (MxK row major) * B[n] (KxNcol) + bias.
// Block (256 thr, 8 waves) owns a 128x32 C tile; per 32-wide K step the block
// stages A (128x32, f32->f16, b128 loads) and B (32x32, stored transposed so
// fragments are contiguous) in LDS; each wave runs 2 WMMAs (16x16 subtiles).
// ---------------------------------------------------------------------------
__global__ void gemm16(const float* __restrict__ A, int lda, long astride,
                       const float* __restrict__ B, int ldb, long bstride,
                       float* __restrict__ C, int ldc, long cstride,
                       const float* __restrict__ bias,
                       int MB, int NBn, int nb, int K) {
  __shared__ __align__(16) _Float16 LA[128 * 32];   // [row][k]
  __shared__ __align__(16) _Float16 LB[32 * 32];    // [ncol][k] (transposed)
  int tid = threadIdx.x, wv = tid >> 5, lane = tid & 31;
  int blk = blockIdx.x;
  int n   = blk / (MB * NBn);
  int rem = blk % (MB * NBn);
  int m0  = (rem / NBn) * 128;
  int t0  = (rem % NBn) * 32;
  int idx = lane & 15, half = lane >> 4;

  const float* Ab = A + (long)n * astride;
  const float* Bb = B + (long)n * bstride;
  float*       Cb = C + (long)n * cstride;

  v8f acc[2];
#pragma unroll
  for (int r = 0; r < 8; ++r) {
    float bi = bias ? bias[m0 + 16 * wv + r + half * 8] : 0.0f;
    acc[0][r] = bi; acc[1][r] = bi;
  }

  int arow = tid >> 1, acol = (tid & 1) * 16;       // A staging: 16 floats/thr
  int brow = tid >> 3, bcol = (tid & 7) * 4;        // B staging: 4 floats/thr

  for (int kb = 0; kb < K; kb += 32) {
    __syncthreads();
    {   // stage A tile
      const float* src = &Ab[(long)(m0 + arow) * lda + kb + acol];
#pragma unroll
      for (int q = 0; q < 2; ++q) {
        float4 f0 = *(const float4*)(src + q * 8);
        float4 f1 = *(const float4*)(src + q * 8 + 4);
        *(h8*)&LA[arow * 32 + acol + q * 8] = cvt8(f0, f1);
      }
    }
    {   // stage B tile (transposed)
      float4 f = *(const float4*)(&Bb[(long)(kb + brow) * ldb + t0 + bcol]);
      LB[(bcol + 0) * 32 + brow] = (_Float16)f.x;
      LB[(bcol + 1) * 32 + brow] = (_Float16)f.y;
      LB[(bcol + 2) * 32 + brow] = (_Float16)f.z;
      LB[(bcol + 3) * 32 + brow] = (_Float16)f.w;
    }
    if (kb + 32 < K) {          // gfx1250 global_prefetch_b8 of next tiles
      __builtin_prefetch(&Ab[(long)(m0 + arow) * lda + kb + 32 + acol], 0, 1);
      __builtin_prefetch(&Bb[(long)(kb + 32 + brow) * ldb + t0 + bcol], 0, 1);
    }
    __syncthreads();

    const _Float16* ap = &LA[(16 * wv + idx) * 32];
    v16h av = frag16(*(const h8*)(ap + half * 8),
                     *(const h8*)(ap + 16 + half * 8));
#pragma unroll
    for (int sub = 0; sub < 2; ++sub) {
      const _Float16* bp = &LB[(sub * 16 + idx) * 32 + half * 8];
      v16h bv = frag16(*(const h8*)bp, *(const h8*)(bp + 16));
      acc[sub] = __builtin_amdgcn_wmma_f32_16x16x32_f16(false, av, false, bv,
                                                        (short)0, acc[sub],
                                                        false, false);
    }
  }

#pragma unroll
  for (int r = 0; r < 8; ++r) {
    int row = m0 + 16 * wv + r + half * 8;
    Cb[(long)row * ldc + t0 + idx]      = acc[0][r];
    Cb[(long)row * ldc + t0 + 16 + idx] = acc[1][r];
  }
}

// ---------------------------------------------------------------------------
// Repack wk [128,256,Ks] -> wkT [Ks][128][256] so A tiles are f32x4-stageable.
// ---------------------------------------------------------------------------
__global__ void repack_wk(const float* __restrict__ wk, float* __restrict__ wkT,
                          int Ks) {
  int idx = blockIdx.x * blockDim.x + threadIdx.x;
  if (idx >= Ks * 128 * 256) return;
  int kk  = idx / (128 * 256);
  int rem = idx % (128 * 256);              // co*256 + ci
  wkT[idx] = wk[(long)rem * Ks + kk];
}

// ---------------------------------------------------------------------------
// region_key_conv: kout[n,co,t] = bk[co] + sum_kk sum_ci wkT[kk,co,ci]*h[n,ci,t+kk-a]
// Block owns full co (128) x 32 cols. Per 32-wide ci step: stage ONE extended
// B tile (width 32+2a, zero-filled, transposed), then for each tap kk stage
// A from wkT[kk] and slice B fragments at column offset +kk from LDS.
// ---------------------------------------------------------------------------
__global__ void region_key_conv(const float* __restrict__ wkT,
                                const float* __restrict__ bk,
                                const float* __restrict__ Hall, int seg,
                                float* __restrict__ kout, int a) {
  const int Ks = 2 * a + 1, W = 32 + 2 * a;
  __shared__ __align__(16) _Float16 LA[128 * 32];   // [co][ci]
  __shared__ __align__(16) _Float16 LB[42 * 32];    // [extcol][ci]
  int tid = threadIdx.x, wv = tid >> 5, lane = tid & 31;
  int n  = blockIdx.x / 6;
  int t0 = (blockIdx.x % 6) * 32;
  int idx = lane & 15, half = lane >> 4;
  const float* hb = Hall + ((long)n * 1024 + seg * 256) * TT;

  v8f acc[2];
#pragma unroll
  for (int r = 0; r < 8; ++r) {
    float bi = bk[16 * wv + r + half * 8];
    acc[0][r] = bi; acc[1][r] = bi;
  }

  int arow = tid >> 1, acol = (tid & 1) * 16;

  for (int cb = 0; cb < 256; cb += 32) {
    for (int kk = 0; kk < Ks; ++kk) {
      __syncthreads();
      if (kk == 0) {        // stage extended B tile once per ci block
        for (int p = tid; p < 32 * W; p += 256) {
          int k = p / W, e = p % W;
          int gcol = t0 - a + e;
          float v = (gcol >= 0 && gcol < TT)
                        ? hb[(long)(cb + k) * TT + gcol] : 0.0f;
          LB[e * 32 + k] = (_Float16)v;
        }
      }
      {                     // stage A tile from wkT[kk]
        const float* src = wkT + (long)kk * 128 * 256
                               + (long)arow * 256 + cb + acol;
#pragma unroll
        for (int q = 0; q < 2; ++q) {
          float4 f0 = *(const float4*)(src + q * 8);
          float4 f1 = *(const float4*)(src + q * 8 + 4);
          *(h8*)&LA[arow * 32 + acol + q * 8] = cvt8(f0, f1);
        }
      }
      __syncthreads();

      const _Float16* ap = &LA[(16 * wv + idx) * 32];
      v16h av = frag16(*(const h8*)(ap + half * 8),
                       *(const h8*)(ap + 16 + half * 8));
#pragma unroll
      for (int sub = 0; sub < 2; ++sub) {
        int e = sub * 16 + idx + kk;                // shifted extended column
        const _Float16* bp = &LB[e * 32 + half * 8];
        v16h bv = frag16(*(const h8*)bp, *(const h8*)(bp + 16));
        acc[sub] = __builtin_amdgcn_wmma_f32_16x16x32_f16(false, av, false, bv,
                                                          (short)0, acc[sub],
                                                          false, false);
      }
    }
  }

#pragma unroll
  for (int r = 0; r < 8; ++r) {
    int row = 16 * wv + r + half * 8;
    kout[((long)n * 128 + row) * TT + t0 + idx]      = acc[0][r];
    kout[((long)n * 128 + row) * TT + t0 + 16 + idx] = acc[1][r];
  }
}

// ---------------------------------------------------------------------------
// Q[n,s,dd,r] = sum_c wa_s[c,dd] * k_s[n,c,r]   (dd padded to 11)
// ---------------------------------------------------------------------------
__global__ void compute_q(const float* __restrict__ kbuf,
                          const float* __restrict__ wa0,
                          const float* __restrict__ wa1,
                          const float* __restrict__ wa2,
                          float* __restrict__ Q) {
  int idx = blockIdx.x * blockDim.x + threadIdx.x;
  if (idx >= NB * 3 * 11 * TT) return;
  int r  = idx % TT;
  int dd = (idx / TT) % 11;
  int s  = (idx / (TT * 11)) % 3;
  int n  = idx / (TT * 11 * 3);
  int a = 2 * s + 1, Ks = 2 * a + 1;                // ADJ = 1,3,5
  if (dd >= Ks) { Q[idx] = 0.0f; return; }
  const float* wa = (s == 0) ? wa0 : (s == 1) ? wa1 : wa2;
  const float* ks = kbuf + (long)s * NB * 128 * TT + (long)n * 128 * TT;
  float acc = 0.0f;
  for (int c = 0; c < 128; ++c)
    acc += wa[c * Ks + dd] * ks[(long)c * TT + r];
  Q[idx] = acc;
}

// ---------------------------------------------------------------------------
// Column softmax over rows, summed over the 3 scales. One block (192 thr =
// 6 wave32) per (n, j); thread i owns row i. Out-of-band logit == ba exactly.
// ---------------------------------------------------------------------------
__device__ inline float block_reduce(float v, float* sm, bool is_max) {
#pragma unroll
  for (int off = 16; off > 0; off >>= 1) {
    float o = __shfl_xor(v, off);
    v = is_max ? fmaxf(v, o) : (v + o);
  }
  __syncthreads();
  if ((threadIdx.x & 31) == 0) sm[threadIdx.x >> 5] = v;
  __syncthreads();
  float r = sm[0];
  for (int w = 1; w < 6; ++w) r = is_max ? fmaxf(r, sm[w]) : (r + sm[w]);
  return r;
}

__global__ void softmax_cols(const float* __restrict__ Q,
                             const float* __restrict__ ba0,
                             const float* __restrict__ ba1,
                             const float* __restrict__ ba2,
                             float* __restrict__ Aout) {
  __shared__ float sm[8];
  int n = blockIdx.x / TT;
  int j = blockIdx.x % TT;
  int i = threadIdx.x;
  float bav[3] = {ba0[0], ba1[0], ba2[0]};

  float aval = 0.0f;
  for (int s = 0; s < 3; ++s) {
    int a = 2 * s + 1, Ks = 2 * a + 1;
    float l = bav[s];
    const float* Qs = Q + (long)((n * 3 + s) * 11) * TT;
    for (int dd = 0; dd < Ks; ++dd) {
      int r = i + dd - a;
      int d = r > j ? r - j : j - r;
      if (r >= 0 && r < TT && d <= a) l += Qs[dd * TT + r];
    }
    float m = block_reduce(l, sm, true);
    float e = __expf(l - m);
    float den = block_reduce(e, sm, false);
    aval += e / den;
  }
  Aout[((long)n * TT + i) * TT + j] = aval;
}

// ---------------------------------------------------------------------------
// Grouped conv (32 groups x 8ch, K=3, pad=1) + ReLU.  h1 = Hall segment 3.
// ---------------------------------------------------------------------------
__global__ void grouped_conv_relu(const float* __restrict__ Hall,
                                  const float* __restrict__ w2,
                                  const float* __restrict__ b2,
                                  float* __restrict__ h2) {
  int idx = blockIdx.x * blockDim.x + threadIdx.x;
  if (idx >= NB * 256 * TT) return;
  int t  = idx % TT;
  int oc = (idx / TT) % 256;
  int n  = idx / (TT * 256);
  const float* hb = Hall + ((long)n * 1024 + 768 + (oc >> 3) * 8) * TT;
  float acc = b2[oc];
#pragma unroll
  for (int kk = 0; kk < 3; ++kk) {
    int tt = t + kk - 1;
    if (tt < 0 || tt >= TT) continue;
#pragma unroll
    for (int jc = 0; jc < 8; ++jc)
      acc += w2[(oc * 8 + jc) * 3 + kk] * hb[(long)jc * TT + tt];
  }
  h2[idx] = fmaxf(acc, 0.0f);
}

// ---------------------------------------------------------------------------
extern "C" void kernel_launch(void* const* d_in, const int* in_sizes, int n_in,
                              void* d_out, int out_size, void* d_ws, size_t ws_size,
                              hipStream_t stream) {
  (void)in_sizes; (void)n_in; (void)out_size; (void)ws_size;
  const float* x  = (const float*)d_in[0];
  const float* wp[3] = {(const float*)d_in[1],  (const float*)d_in[7],  (const float*)d_in[13]};
  const float* bp[3] = {(const float*)d_in[2],  (const float*)d_in[8],  (const float*)d_in[14]};
  const float* wk[3] = {(const float*)d_in[3],  (const float*)d_in[9],  (const float*)d_in[15]};
  const float* bk[3] = {(const float*)d_in[4],  (const float*)d_in[10], (const float*)d_in[16]};
  const float* wa[3] = {(const float*)d_in[5],  (const float*)d_in[11], (const float*)d_in[17]};
  const float* ba[3] = {(const float*)d_in[6],  (const float*)d_in[12], (const float*)d_in[18]};
  const float* w1 = (const float*)d_in[19];
  const float* b1 = (const float*)d_in[20];
  const float* w2 = (const float*)d_in[21];
  const float* b2 = (const float*)d_in[22];
  const float* w3 = (const float*)d_in[23];
  const float* b3 = (const float*)d_in[24];

  // Workspace layout (floats). Total ~14.8 MB.
  float* ws   = (float*)d_ws;
  float* Hall = ws;                    // [8][1024][192]  = 1,572,864
  float* kbuf = Hall + 1572864;        // 3*[8][128][192] =   589,824
  float* Q    = kbuf + 589824;         // [8][3][11][192] =    50,688
  float* h2   = Q + 50688;             // [8][256][192]   =   393,216
  float* h3   = h2 + 393216;           // [8][256][192]   =   393,216
  float* wkT[3];
  wkT[0] = h3 + 393216;                // [3][128][256]   =    98,304
  wkT[1] = wkT[0] + 3 * 128 * 256;     // [7][128][256]   =   229,376
  wkT[2] = wkT[1] + 7 * 128 * 256;     // [11][128][256]  =   360,448

  float* out1 = (float*)d_out;                 // xs @ A : 8*256*192
  float* Amat = out1 + NB * 256 * TT;          // A      : 8*192*192

  const long xbs = 512L * TT, hbs = 1024L * TT, cbs = 256L * TT;
  const int GBLK = 2 * 6 * NB;                 // 96 blocks per GEMM

  // Stage 0: repack region-key weights for vectorized staging.
  for (int s = 0; s < 3; ++s) {
    int Ks = 2 * (2 * s + 1) + 1;
    repack_wk<<<(Ks * 128 * 256 + 255) / 256, 256, 0, stream>>>(wk[s], wkT[s], Ks);
  }

  // Stage 1: stacked 1x1 projections (wp0, wp1, wp2, w1) -> Hall segments.
  for (int s = 0; s < 3; ++s)
    gemm16<<<GBLK, 256, 0, stream>>>(wp[s], 512, 0, x, TT, xbs,
                                     Hall + (long)s * 256 * TT, TT, hbs,
                                     bp[s], 2, 6, NB, 512);
  gemm16<<<GBLK, 256, 0, stream>>>(w1, 512, 0, x, TT, xbs,
                                   Hall + 3L * 256 * TT, TT, hbs,
                                   b1, 2, 6, NB, 512);

  // Stage 2: region key convs (K-tap, WMMA, extended LDS tile per ci block).
  for (int s = 0; s < 3; ++s)
    region_key_conv<<<NB * 6, 256, 0, stream>>>(wkT[s], bk[s], Hall, s,
                                                kbuf + (long)s * NB * 128 * TT,
                                                2 * s + 1);

  // Stage 3: Q = wa . k  (per tap, per row).
  compute_q<<<(NB * 3 * 11 * TT + 255) / 256, 256, 0, stream>>>(
      kbuf, wa[0], wa[1], wa[2], Q);

  // Stage 4: banded logits + column softmax, summed over scales -> A.
  softmax_cols<<<NB * TT, TT, 0, stream>>>(Q, ba[0], ba[1], ba[2], Amat);

  // Stage 5: grouped conv + relu on h1 (Hall segment 3).
  grouped_conv_relu<<<(NB * 256 * TT + 255) / 256, 256, 0, stream>>>(
      Hall, w2, b2, h2);

  // Stage 6: h3 = w3 @ h2 + b3.
  gemm16<<<GBLK, 256, 0, stream>>>(w3, 256, 0, h2, TT, cbs,
                                   h3, TT, cbs, b3, 2, 6, NB, 256);

  // Stage 7: out = h3 @ A  (K = 192).
  gemm16<<<GBLK, 256, 0, stream>>>(h3, TT, cbs, Amat, TT, (long)TT * TT,
                                   out1, TT, cbs, nullptr, 2, 6, NB, 192);
}
